// SelfAttention_79637283602810
// MI455X (gfx1250) — matrix operand order repaired and checked by reading, not compiled, not run on hardware
//
#include <hip/hip_runtime.h>

typedef __bf16 bf16_t;
typedef __attribute__((ext_vector_type(16))) __bf16 v16bf;
typedef __attribute__((ext_vector_type(8)))  __bf16 v8bf;
typedef __attribute__((ext_vector_type(4)))  __bf16 v4bf;
typedef __attribute__((ext_vector_type(8)))  float  v8f;

#define BATCH 8
#define SEQ   2048
#define DIM   512

// ---------------------------------------------------------------------------
// WMMA helpers (CDNA5 wave32, 16x16x32 bf16 -> f32 accumulate)
// ---------------------------------------------------------------------------
__device__ inline v8f wmma_bf16(v16bf a, v16bf b, v8f c) {
  return __builtin_amdgcn_wmma_f32_16x16x32_bf16(
      /*neg_a=*/false, a, /*neg_b=*/false, b,
      /*c_mod=*/(short)0, c, /*reuse_a=*/false, /*reuse_b=*/false);
}

// A fragment: 16x32 (MxK), row-major source with leading dim `ld` (elements).
// Lane L (<16): M=L, K = 0..15 ; lane L (>=16): M=L-16, K = 16..31.
__device__ inline v16bf load_a_frag(const bf16_t* src, int ld) {
  const int lane = threadIdx.x & 31;
  const int m    = lane & 15;
  const int kb   = (lane >> 4) * 16;
  const bf16_t* p = src + (size_t)m * ld + kb;
  v8bf lo = *(const v8bf*)(p);
  v8bf hi = *(const v8bf*)(p + 8);
  v16bf a;
#pragma unroll
  for (int i = 0; i < 8; ++i) { a[i] = lo[i]; a[8 + i] = hi[i]; }
  return a;
}

// B fragment: 32x16 (KxN) where B[k][n] = src[n*ld + k]  (src is the N-major
// matrix whose rows are contracted: weight[e][d], K[key][d], Vt[d][key]).
// Lane L: N = L&15, K = (L>>4)*16 .. +15, contiguous 32B in memory.
__device__ inline v16bf load_bT_frag(const bf16_t* src, int ld) {
  const int lane = threadIdx.x & 31;
  const int n    = lane & 15;
  const int kb   = (lane >> 4) * 16;
  const bf16_t* p = src + (size_t)n * ld + kb;
  v8bf lo = *(const v8bf*)(p);
  v8bf hi = *(const v8bf*)(p + 8);
  v16bf b;
#pragma unroll
  for (int i = 0; i < 8; ++i) { b[i] = lo[i]; b[8 + i] = hi[i]; }
  return b;
}

// ---------------------------------------------------------------------------
// Kernel 0: fp32 -> bf16 conversion (vector-4)
// ---------------------------------------------------------------------------
__global__ void cvt_f32_to_bf16(const float* __restrict__ src,
                                bf16_t* __restrict__ dst, int n4) {
  int i = blockIdx.x * blockDim.x + threadIdx.x;
  if (i < n4) {
    float4 f = ((const float4*)src)[i];
    v4bf o;
    o[0] = (bf16_t)f.x; o[1] = (bf16_t)f.y;
    o[2] = (bf16_t)f.z; o[3] = (bf16_t)f.w;
    *(v4bf*)(dst + (size_t)i * 4) = o;
  }
}

// ---------------------------------------------------------------------------
// Kernel 1: QKV projection.  out[m,e] = sum_d X[m,d]*W[e,d] + b[e].
// z==0 -> K (row-major [b*SEQ+n][e]), z==1 -> Q (row-major),
// z==2 -> V stored TRANSPOSED: Vt[b][e][n]  (so P@V B-frags are contiguous).
// Block = 8 waves; wave tile = 16x64; block tile = 128 rows x 64 cols.
// grid = (M/128, DIM/64, 3)
// ---------------------------------------------------------------------------
__global__ __launch_bounds__(256) void qkv_proj(
    const bf16_t* __restrict__ Xb,
    const bf16_t* __restrict__ Wkb, const bf16_t* __restrict__ Wqb,
    const bf16_t* __restrict__ Wvb,
    const float* __restrict__ bk, const float* __restrict__ bq,
    const float* __restrict__ bv,
    bf16_t* __restrict__ Kb, bf16_t* __restrict__ Qb, bf16_t* __restrict__ Vt) {
  const int w    = threadIdx.x >> 5;
  const int lane = threadIdx.x & 31;
  const int m0   = blockIdx.x * 128 + w * 16;
  const int n0   = blockIdx.y * 64;
  const int z    = blockIdx.z;

  const bf16_t* W    = (z == 0) ? Wkb : (z == 1) ? Wqb : Wvb;
  const float*  bias = (z == 0) ? bk  : (z == 1) ? bq  : bv;

  v8f acc[4];
#pragma unroll
  for (int j = 0; j < 4; ++j) acc[j] = (v8f){0, 0, 0, 0, 0, 0, 0, 0};

  for (int kb = 0; kb < DIM; kb += 32) {
    v16bf a = load_a_frag(Xb + (size_t)m0 * DIM + kb, DIM);
#pragma unroll
    for (int j = 0; j < 4; ++j) {
      v16bf bm = load_bT_frag(W + (size_t)(n0 + j * 16) * DIM + kb, DIM);
      acc[j] = wmma_bf16(a, bm, acc[j]);
    }
  }

  const int nI = lane & 15;
  const int hi = lane >> 4;
  if (z == 2) {
    // transposed store: lane's 8 rows are consecutive seq positions -> 16B store
    const int bb = m0 / SEQ;
    const int ml = m0 % SEQ;
#pragma unroll
    for (int j = 0; j < 4; ++j) {
      float bvv = bias[n0 + j * 16 + nI];
      v8bf col;
#pragma unroll
      for (int v = 0; v < 8; ++v) col[v] = (bf16_t)(acc[j][v] + bvv);
      bf16_t* d0 = Vt + ((size_t)bb * DIM + (n0 + j * 16 + nI)) * SEQ + ml + hi * 8;
      *(v8bf*)d0 = col;
    }
  } else {
    bf16_t* dst = (z == 0) ? Kb : Qb;
#pragma unroll
    for (int j = 0; j < 4; ++j) {
      float bvv = bias[n0 + j * 16 + nI];
#pragma unroll
      for (int v = 0; v < 8; ++v) {
        int r = hi * 8 + v;
        dst[(size_t)(m0 + r) * DIM + n0 + j * 16 + nI] = (bf16_t)(acc[j][v] + bvv);
      }
    }
  }
}

// ---------------------------------------------------------------------------
// Kernel 2: flash attention.  One block per (16-query tile, batch).
// 8 waves; each wave computes one 16-key score slice of the 128-key block,
// then owns 64 output columns for P@V.  Online softmax, f32 accumulators.
// grid = (SEQ/16, BATCH)
// ---------------------------------------------------------------------------
__global__ __launch_bounds__(256) void flash_attn(
    const bf16_t* __restrict__ Qb, const bf16_t* __restrict__ Kb,
    const bf16_t* __restrict__ Vt, const int* __restrict__ maskp,
    float* __restrict__ out) {
  __shared__ bf16_t Qs[16 * DIM];     // 16 KB: query tile
  __shared__ bf16_t Ps[16 * 128];     // 4 KB: softmaxed probs for key block
  __shared__ float  redmax[8 * 16];
  __shared__ float  redsum[8 * 16];

  const int b    = blockIdx.y;
  const int q0   = blockIdx.x * 16;
  const int tid  = threadIdx.x;
  const int w    = tid >> 5;
  const int lane = tid & 31;
  const int nI   = lane & 15;
  const int hi   = lane >> 4;
  const int c0   = w * 64;            // this wave's output column base
  const float scale = 0.044194173824159216f;  // 1/sqrt(512)

  // Stage Q tile in LDS via CDNA5 async global->LDS copy (bypasses VGPRs).
  // 16 rows x 512 bf16 = 16384 B = 1024 x 16B chunks; 4 chunks per thread.
  {
    const char* gbase = (const char*)(Qb + (size_t)(b * SEQ + q0) * DIM);
    uint32_t lds_base = (uint32_t)(uintptr_t)(&Qs[0]);
    for (int i = tid; i < (16 * DIM * 2) / 16; i += 256) {
      uint32_t ldsa = lds_base + (uint32_t)i * 16u;
      unsigned long long ga = (unsigned long long)(gbase + (size_t)i * 16);
      asm volatile("global_load_async_to_lds_b128 %0, %1, off"
                   :: "v"(ldsa), "v"(ga) : "memory");
    }
    asm volatile("s_wait_asynccnt 0" ::: "memory");
  }
  __syncthreads();

  float m_run[8], l_run[8];
  v8f acc[4];
#pragma unroll
  for (int v = 0; v < 8; ++v) { m_run[v] = -3.0e38f; l_run[v] = 0.f; }
#pragma unroll
  for (int j = 0; j < 4; ++j) acc[j] = (v8f){0, 0, 0, 0, 0, 0, 0, 0};

  for (int kb0 = 0; kb0 < SEQ; kb0 += 128) {
    const int kcol = kb0 + w * 16;

    // prefetch next key block's K rows for this wave (stream is predictable)
    if (kb0 + 128 < SEQ) {
      const bf16_t* nk = Kb + (size_t)(b * SEQ + kcol + 128) * DIM;
      __builtin_prefetch(nk + (size_t)nI * DIM + (lane >> 4) * 256, 0, 1);
    }

    // ---- scores: S = Q(16xD) @ K[kcol..kcol+16]^T, this wave's 16 keys ----
    v8f s = (v8f){0, 0, 0, 0, 0, 0, 0, 0};
    for (int kk = 0; kk < DIM; kk += 32) {
      v16bf a  = load_a_frag(Qs + kk, DIM);
      v16bf bm = load_bT_frag(Kb + (size_t)(b * SEQ + kcol) * DIM + kk, DIM);
      s = wmma_bf16(a, bm, s);
    }

    // scale + mask (lane nI owns key column kcol+nI)
    const bool live = maskp[b * SEQ + kcol + nI] != 0;
    float sv[8];
#pragma unroll
    for (int v = 0; v < 8; ++v) {
      float x = s[v] * scale;
      sv[v] = live ? x : -1.0e9f;
    }

    // wave row-max across the 16 columns (xor masks stay inside each half)
    float rm[8];
#pragma unroll
    for (int v = 0; v < 8; ++v) rm[v] = sv[v];
    for (int xm = 1; xm < 16; xm <<= 1) {
#pragma unroll
      for (int v = 0; v < 8; ++v)
        rm[v] = fmaxf(rm[v], __shfl_xor(rm[v], xm, 32));
    }
    if (nI == 0) {
#pragma unroll
      for (int v = 0; v < 8; ++v) redmax[w * 16 + hi * 8 + v] = rm[v];
    }
    __syncthreads();

    // block max + online-softmax correction (identical in all lanes per row)
    float mnew[8], alpha[8];
#pragma unroll
    for (int v = 0; v < 8; ++v) {
      int r = hi * 8 + v;
      float bm2 = redmax[r];
      for (int w2 = 1; w2 < 8; ++w2) bm2 = fmaxf(bm2, redmax[w2 * 16 + r]);
      mnew[v]  = fmaxf(m_run[v], bm2);
      alpha[v] = __expf(m_run[v] - mnew[v]);
    }

    // P = exp(S - mnew), wave row-sum
    float p[8], rs[8];
#pragma unroll
    for (int v = 0; v < 8; ++v) { p[v] = __expf(sv[v] - mnew[v]); rs[v] = p[v]; }
    for (int xm = 1; xm < 16; xm <<= 1) {
#pragma unroll
      for (int v = 0; v < 8; ++v) rs[v] += __shfl_xor(rs[v], xm, 32);
    }

    // stash P tile (bf16) and partial sums
#pragma unroll
    for (int v = 0; v < 8; ++v) {
      int r = hi * 8 + v;
      Ps[r * 128 + w * 16 + nI] = (bf16_t)p[v];
    }
    if (nI == 0) {
#pragma unroll
      for (int v = 0; v < 8; ++v) redsum[w * 16 + hi * 8 + v] = rs[v];
    }
    __syncthreads();

    // update running stats, rescale accumulators
#pragma unroll
    for (int v = 0; v < 8; ++v) {
      int r = hi * 8 + v;
      float bs = 0.f;
      for (int w2 = 0; w2 < 8; ++w2) bs += redsum[w2 * 16 + r];
      l_run[v] = l_run[v] * alpha[v] + bs;
      m_run[v] = mnew[v];
    }
#pragma unroll
    for (int j = 0; j < 4; ++j)
#pragma unroll
      for (int v = 0; v < 8; ++v) acc[j][v] *= alpha[v];

    // ---- P(16x128) @ V(128xD), this wave's 64 columns ----
    // B[k][n] = V[key=kb0+kk2*32+k][dcol=c0+j*16+n] = Vt[dcol][key]: contiguous.
    for (int kk2 = 0; kk2 < 4; ++kk2) {
      v16bf a = load_a_frag(Ps + kk2 * 32, 128);
#pragma unroll
      for (int j = 0; j < 4; ++j) {
        v16bf bm = load_bT_frag(
            Vt + ((size_t)b * DIM + c0 + j * 16) * SEQ + kb0 + kk2 * 32, SEQ);
        acc[j] = wmma_bf16(a, bm, acc[j]);
      }
    }
    __syncthreads();  // protect Ps/redmax/redsum before next key block
  }

  // epilogue: normalize and store fp32
#pragma unroll
  for (int j = 0; j < 4; ++j) {
#pragma unroll
    for (int v = 0; v < 8; ++v) {
      int r = hi * 8 + v;
      float o = acc[j][v] / l_run[v];
      out[(size_t)(b * SEQ + q0 + r) * DIM + c0 + j * 16 + nI] = o;
    }
  }
}

// ---------------------------------------------------------------------------
// Host launcher
// ---------------------------------------------------------------------------
extern "C" void kernel_launch(void* const* d_in, const int* in_sizes, int n_in,
                              void* d_out, int out_size, void* d_ws,
                              size_t ws_size, hipStream_t stream) {
  const float* x  = (const float*)d_in[0];
  const int*   mk = (const int*)d_in[1];
  const float* Wk = (const float*)d_in[2];
  const float* bk = (const float*)d_in[3];
  const float* Wq = (const float*)d_in[4];
  const float* bq = (const float*)d_in[5];
  const float* Wv = (const float*)d_in[6];
  const float* bv = (const float*)d_in[7];
  float* out = (float*)d_out;

  const size_t nX = (size_t)BATCH * SEQ * DIM;  // 8,388,608
  const size_t nW = (size_t)DIM * DIM;          // 262,144

  bf16_t* p   = (bf16_t*)d_ws;
  bf16_t* Xb  = p; p += nX;
  bf16_t* Wkb = p; p += nW;
  bf16_t* Wqb = p; p += nW;
  bf16_t* Wvb = p; p += nW;
  bf16_t* Kb  = p; p += nX;
  bf16_t* Qb  = p; p += nX;
  bf16_t* Vt  = p; p += nX;   // transposed: [BATCH][DIM][SEQ]

  const int tpb = 256;
  cvt_f32_to_bf16<<<(int)((nX / 4 + tpb - 1) / tpb), tpb, 0, stream>>>(x, Xb, (int)(nX / 4));
  cvt_f32_to_bf16<<<(int)((nW / 4 + tpb - 1) / tpb), tpb, 0, stream>>>(Wk, Wkb, (int)(nW / 4));
  cvt_f32_to_bf16<<<(int)((nW / 4 + tpb - 1) / tpb), tpb, 0, stream>>>(Wq, Wqb, (int)(nW / 4));
  cvt_f32_to_bf16<<<(int)((nW / 4 + tpb - 1) / tpb), tpb, 0, stream>>>(Wv, Wvb, (int)(nW / 4));

  qkv_proj<<<dim3((BATCH * SEQ) / 128, DIM / 64, 3), 256, 0, stream>>>(
      Xb, Wkb, Wqb, Wvb, bk, bq, bv, Kb, Qb, Vt);

  flash_attn<<<dim3(SEQ / 16, BATCH), 256, 0, stream>>>(Qb, Kb, Vt, mk, out);
}
